// DGMBlock_18141941858949
// MI455X (gfx1250) — matrix operand order repaired and checked by reading, not compiled, not run on hardware
//
#include <hip/hip_runtime.h>

// Problem constants (match reference)
#define NN 10000
#define EE 160000
#define DD 128
#define CC 128
#define KTOP 4
#define NSTRIP 625   // 10000 / 16 row strips
#define NCH 8        // column chunks for h@h^T
#define TPC 79       // 16-col tiles per chunk (ceil(625/8)); last chunk clipped

typedef unsigned short ushort_t;
typedef __attribute__((ext_vector_type(16))) __bf16 v16bf;
typedef __attribute__((ext_vector_type(8)))  float  v8f;
typedef __attribute__((ext_vector_type(4)))  unsigned v4u;
typedef __attribute__((ext_vector_type(8)))  int    v8i_t;
typedef __attribute__((ext_vector_type(4)))  int    v4i_t;

#if defined(__has_builtin)
#if __has_builtin(__builtin_amdgcn_tensor_load_to_lds) && \
    __has_builtin(__builtin_amdgcn_s_wait_tensorcnt)
#define HAVE_TDM 1
#endif
#endif
#ifndef HAVE_TDM
#define HAVE_TDM 0
#endif

union BfVec { v16bf v; ushort_t u[16]; };

__device__ __forceinline__ ushort_t f2bf(float f) {
  unsigned u = __float_as_uint(f);
  unsigned r = (u + 0x7FFFu + ((u >> 16) & 1u)) >> 16;  // round-to-nearest-even
  return (ushort_t)r;
}

// A fragment (16x32 bf16, M x K), row-major source with leading dim `ld` (halfs).
// Contiguous-k layout (matches load_b_frag): lanes 0-15 -> M=lane, slots 0-15 = K 0-15;
//             lanes 16-31 -> M=lane-16, K 16-31.
__device__ __forceinline__ v16bf load_a_frag(const ushort_t* base, int ld,
                                             int lane, int row0, int k0) {
  int m = lane & 15, hi = lane >> 4;
  const ushort_t* p = base + (size_t)(row0 + m) * ld + k0;
  BfVec r;
#pragma unroll
  for (int i = 0; i < 8; ++i) r.u[i]     = p[(hi ? 16 : 0)  + i];
#pragma unroll
  for (int i = 0; i < 8; ++i) r.u[8 + i] = p[(hi ? 24 : 8) + i];
  return r.v;
}

// B fragment (32x16 bf16, K x N) from a K-contiguous source: src[(col)*ld + k].
// Contiguous-k layout: lanes 0-15 -> N=lane, K k0..k0+15; lanes 16-31 -> N=lane-16, K k0+16..31.
__device__ __forceinline__ v16bf load_b_frag(const ushort_t* baseT, int ld,
                                             int lane, int col0, int k0) {
  int n = lane & 15, hi = lane >> 4;
  const ushort_t* p = baseT + (size_t)(col0 + n) * ld + k0 + (hi ? 16 : 0);
  BfVec r;
#pragma unroll
  for (int i = 0; i < 16; ++i) r.u[i] = p[i];
  return r.v;
}

// Sorted-descending top-4 insert; strict '>' keeps earliest index on ties.
__device__ __forceinline__ void ins4(float v, int ix, float* tv, int* ti) {
  if (v > tv[3]) {
    if (v > tv[2]) {
      tv[3] = tv[2]; ti[3] = ti[2];
      if (v > tv[1]) {
        tv[2] = tv[1]; ti[2] = ti[1];
        if (v > tv[0]) { tv[1] = tv[0]; ti[1] = ti[0]; tv[0] = v; ti[0] = ix; }
        else           { tv[1] = v; ti[1] = ix; }
      } else           { tv[2] = v; ti[2] = ix; }
    } else             { tv[3] = v; ti[3] = ix; }
  }
}

// Issue a TDM 2-D tile load: 16 rows x 16 cols of f32 from q (row stride NN)
// into LDS at byte address lds_addr. Descriptor per CDNA5 ISA sec 8.3/8.4.
__device__ __forceinline__ void tdm_load_q_tile(const float* gsrc, unsigned lds_addr) {
#if HAVE_TDM
  unsigned long long ga = (unsigned long long)(uintptr_t)gsrc;
  v4u g0;
  g0[0] = 1u;                                     // count=1, user descriptor
  g0[1] = lds_addr;                               // lds_addr (bytes)
  g0[2] = (unsigned)ga;                           // global_addr[31:0]
  g0[3] = (unsigned)((ga >> 32) & 0x01FFFFFFu)    // global_addr[56:32]
          | 0x80000000u;                          // type=2 ("image")
  v8i_t g1;
  g1[0] = (int)(2u << 16);                        // data_size=2 -> 4 bytes
  g1[1] = (int)(16u << 16);                       // tensor_dim0 = 16 (low16)
  g1[2] = (int)(16u << 16);                       // tensor_dim1 = 16 (low16)
  g1[3] = (int)(16u << 16);                       // tile_dim0 = 16
  g1[4] = 16;                                     // tile_dim1 = 16, tile_dim2 = 0
  g1[5] = (int)(unsigned)NN;                      // tensor_dim0_stride = 10000
  g1[6] = 0;                                      // stride0 hi, dim1_stride lo
  g1[7] = 0;
  v4i_t gz = {0, 0, 0, 0};
#if __clang_major__ >= 23
  v8i_t gz8 = {0, 0, 0, 0, 0, 0, 0, 0};
  __builtin_amdgcn_tensor_load_to_lds(g0, g1, gz, gz, gz8, 0);
#else
  __builtin_amdgcn_tensor_load_to_lds(g0, g1, gz, gz, 0);
#endif
#endif
}

// ---------------- GCN pipeline ----------------

__global__ void k_conv_x(const float* x, ushort_t* xbf) {
  size_t t = (size_t)blockIdx.x * blockDim.x + threadIdx.x;
  if (t < (size_t)NN * DD) xbf[t] = f2bf(x[t]);
}

__global__ void k_conv_wt(const float* W, ushort_t* wtbf) {
  int t = blockIdx.x * blockDim.x + threadIdx.x;   // t = n*128 + k
  if (t < DD * CC) {
    int n = t >> 7, k = t & 127;
    wtbf[t] = f2bf(W[k * CC + n]);                 // transpose to K-contiguous
  }
}

__global__ void k_init_deg(float* deg) {
  int i = blockIdx.x * blockDim.x + threadIdx.x;
  if (i < NN) deg[i] = 1.0f;                       // self loop
}

__global__ void k_deg(const int* ei, float* deg) {
  int e = blockIdx.x * blockDim.x + threadIdx.x;
  if (e < EE) atomicAdd(&deg[ei[e]], 1.0f);        // rows = ei[0][:]
}

__global__ void k_dinv(const float* deg, float* dinv) {
  int i = blockIdx.x * blockDim.x + threadIdx.x;
  if (i < NN) dinv[i] = rsqrtf(fmaxf(deg[i], 1e-12f));
}

// h0 = x @ W via bf16 WMMA. grid = (625 row strips, 8 col tiles), 32 thr/blk.
__global__ void __launch_bounds__(32)
k_gemm_xw(const ushort_t* xbf, const ushort_t* wtbf, float* h0) {
  int lane = threadIdx.x;
  int i0 = blockIdx.x * 16, n0 = blockIdx.y * 16;
  v8f c = {};
#pragma unroll
  for (int k = 0; k < DD; k += 32) {
    v16bf a = load_a_frag(xbf, DD, lane, i0, k);
    v16bf b = load_b_frag(wtbf, DD, lane, n0, k);
    c = __builtin_amdgcn_wmma_f32_16x16x32_bf16(false, a, false, b,
                                                (short)0, c, false, false);
  }
  int hi = lane >> 4, n = lane & 15;
#pragma unroll
  for (int v = 0; v < 8; ++v)
    h0[(size_t)(i0 + v + (hi ? 8 : 0)) * CC + n0 + n] = c[v];
}

// h = h0 * dinv^2 + b  (self-loop term; initializes output h region)
__global__ void k_self(const float* h0, const float* dinv, const float* b,
                       float* hout) {
  size_t t = (size_t)blockIdx.x * blockDim.x + threadIdx.x;
  if (t < (size_t)NN * CC) {
    int r = (int)(t >> 7), c = (int)(t & 127);
    float di = dinv[r];
    hout[t] = h0[t] * di * di + b[c];
  }
}

// scatter-add edge messages: hout[row] += h0[col] * dinv[row]*dinv[col]
__global__ void k_edges(const int* ei, const float* h0, const float* dinv,
                        float* hout) {
  size_t tot = (size_t)EE * CC;
  size_t stride = (size_t)gridDim.x * blockDim.x;
  for (size_t t = (size_t)blockIdx.x * blockDim.x + threadIdx.x; t < tot;
       t += stride) {
    int e = (int)(t >> 7), c = (int)(t & 127);
    int r  = ei[e];
    int cl = ei[EE + e];
    float w = dinv[r] * dinv[cl];
    atomicAdd(&hout[(size_t)r * CC + c], h0[(size_t)cl * CC + c] * w);
  }
}

// per-row squared norm + bf16 conversion of h
__global__ void __launch_bounds__(128)
k_sq_bf(const float* h, ushort_t* hbf, float* sq) {
  int r = blockIdx.x, c = threadIdx.x;
  float v = h[(size_t)r * CC + c];
  hbf[(size_t)r * CC + c] = f2bf(v);
  __shared__ float red[128];
  red[c] = v * v;
  __syncthreads();
  for (int s = 64; s > 0; s >>= 1) {
    if (c < s) red[c] += red[c + s];
    __syncthreads();
  }
  if (c == 0) sq[r] = red[0];
}

// ---------------- fused h@h^T + gumbel + partial top-4 ----------------
// grid = (625 row strips, 8 column chunks), 32 threads (one wave) per block.
// q tiles are streamed through LDS by the Tensor Data Mover, double-buffered,
// overlapped with the WMMA + gumbel compute of the previous tile.
__global__ void __launch_bounds__(32)
k_hht_topk(const ushort_t* hbf, const float* sq, const float* q,
           const float* tempp, float* pval, int* pidx) {
  int s = blockIdx.x, ch = blockIdx.y, lane = threadIdx.x;
  int i0 = s * 16;
  int t0 = ch * TPC;
  int t1 = t0 + TPC; if (t1 > NSTRIP) t1 = NSTRIP;
  float T = tempp[0];

  __shared__ float qtile[2][16 * 16];   // TDM-loaded q tiles (double buffer)
  __shared__ float stage[16][17];       // lq tile transpose staging
  __shared__ float mval[16][KTOP];      // half-wave top-4 merge
  __shared__ int   midx[16][KTOP];

#if HAVE_TDM
  unsigned qb[2];
  qb[0] = (unsigned)(uintptr_t)&qtile[0][0];
  qb[1] = (unsigned)(uintptr_t)&qtile[1][0];
#endif

  // A fragments for this 16-row strip, reused across the whole column chunk
  v16bf a0 = load_a_frag(hbf, CC, lane, i0, 0);
  v16bf a1 = load_a_frag(hbf, CC, lane, i0, 32);
  v16bf a2 = load_a_frag(hbf, CC, lane, i0, 64);
  v16bf a3 = load_a_frag(hbf, CC, lane, i0, 96);

  int hi = lane >> 4, n = lane & 15;
  float sqa[8];
#pragma unroll
  for (int v = 0; v < 8; ++v) sqa[v] = sq[i0 + v + (hi ? 8 : 0)];

  float tv[KTOP] = {-INFINITY, -INFINITY, -INFINITY, -INFINITY};
  int   ti[KTOP] = {0, 0, 0, 0};

#if HAVE_TDM
  tdm_load_q_tile(q + (size_t)i0 * NN + (size_t)t0 * 16, qb[0]);   // prime pipe
#endif

  for (int t = t0; t < t1; ++t) {
    int j0 = t * 16;
    int buf = (t - t0) & 1;

#if HAVE_TDM
    if (t + 1 < t1)   // prefetch next q tile while we compute this one
      tdm_load_q_tile(q + (size_t)i0 * NN + (size_t)(t + 1) * 16, qb[buf ^ 1]);
#endif

    v8f c = {};
    c = __builtin_amdgcn_wmma_f32_16x16x32_bf16(false, a0, false,
          load_b_frag(hbf, CC, lane, j0, 0),  (short)0, c, false, false);
    c = __builtin_amdgcn_wmma_f32_16x16x32_bf16(false, a1, false,
          load_b_frag(hbf, CC, lane, j0, 32), (short)0, c, false, false);
    c = __builtin_amdgcn_wmma_f32_16x16x32_bf16(false, a2, false,
          load_b_frag(hbf, CC, lane, j0, 64), (short)0, c, false, false);
    c = __builtin_amdgcn_wmma_f32_16x16x32_bf16(false, a3, false,
          load_b_frag(hbf, CC, lane, j0, 96), (short)0, c, false, false);

#if HAVE_TDM
    if (t + 1 < t1) __builtin_amdgcn_s_wait_tensorcnt(1);  // current tile done
    else            __builtin_amdgcn_s_wait_tensorcnt(0);
    asm volatile("" ::: "memory");
#endif

    float sqb = sq[j0 + n];
#pragma unroll
    for (int v = 0; v < 8; ++v) {
      int row = v + (hi ? 8 : 0);
      float d2 = fmaxf(sqa[v] + sqb - 2.0f * c[v], 0.0f);
#if HAVE_TDM
      float qv = qtile[buf][row * 16 + n];
#else
      float qv = q[(size_t)(i0 + row) * NN + (j0 + n)];
#endif
      float g  = -__logf(-__logf(qv + 1e-8f));   // gumbel from U(0,1)
      stage[row][n] = -T * d2 + g;
    }
    __syncthreads();
    // all 32 lanes scan: lane owns row lane%16, columns hi*8 .. hi*8+7
#pragma unroll
    for (int u = 0; u < 8; ++u) {
      int col = (hi << 3) + u;
      ins4(stage[n][col], j0 + col, tv, ti);
    }
    __syncthreads();
  }

  // merge the two half-wave top-4 lists per row via LDS
  if (lane >= 16) {
#pragma unroll
    for (int t = 0; t < KTOP; ++t) { mval[n][t] = tv[t]; midx[n][t] = ti[t]; }
  }
  __syncthreads();
  if (lane < 16) {
#pragma unroll
    for (int t = 0; t < KTOP; ++t) ins4(mval[n][t], midx[n][t], tv, ti);
    size_t base = ((size_t)(s * NCH + ch) * 16 + n) * KTOP;
#pragma unroll
    for (int t = 0; t < KTOP; ++t) { pval[base + t] = tv[t]; pidx[base + t] = ti[t]; }
  }
}

// merge 8 per-chunk top-4 lists per row; emit edges + logprobs + self loops
__global__ void k_merge(const float* pval, const int* pidx, float* out) {
  int r = blockIdx.x * blockDim.x + threadIdx.x;
  if (r >= NN) return;
  int s = r >> 4, row = r & 15;
  float tv[KTOP] = {-INFINITY, -INFINITY, -INFINITY, -INFINITY};
  int   ti[KTOP] = {0, 0, 0, 0};
  for (int ch = 0; ch < NCH; ++ch) {          // ascending-j order preserves ties
    size_t base = ((size_t)(s * NCH + ch) * 16 + row) * KTOP;
#pragma unroll
    for (int t = 0; t < KTOP; ++t) ins4(pval[base + t], pidx[base + t], tv, ti);
  }
  float* EH = out + (size_t)NN * CC;          // edge_index_hat [2, 50000] as float
  const int M = NN * KTOP + NN;               // 50000
  float* LP = EH + 2 * M;                     // logprobs [NN, 4]
#pragma unroll
  for (int t = 0; t < KTOP; ++t) {
    EH[r * KTOP + t]     = (float)ti[t];      // cols = topk idx
    EH[M + r * KTOP + t] = (float)r;          // rows
    LP[r * KTOP + t]     = tv[t];
  }
  EH[NN * KTOP + r]     = (float)r;           // appended self loops
  EH[M + NN * KTOP + r] = (float)r;
}

// ---------------- launch ----------------
extern "C" void kernel_launch(void* const* d_in, const int* in_sizes, int n_in,
                              void* d_out, int out_size, void* d_ws, size_t ws_size,
                              hipStream_t stream) {
  const float* x    = (const float*)d_in[0];
  const int*   ei   = (const int*)d_in[1];     // [2, E] flat
  const float* q    = (const float*)d_in[2];   // [N, N]
  const float* W    = (const float*)d_in[3];   // [D, C]
  const float* b    = (const float*)d_in[4];   // [C]
  const float* temp = (const float*)d_in[5];   // scalar

  // workspace carve-up (~13 MB)
  char* w = (char*)d_ws;
  ushort_t* xbf  = (ushort_t*)w;                       w += (size_t)NN * DD * 2;
  ushort_t* wtbf = (ushort_t*)w;                       w += (size_t)DD * CC * 2;
  ushort_t* hbf  = (ushort_t*)w;                       w += (size_t)NN * CC * 2;
  float*    h0   = (float*)w;                          w += (size_t)NN * CC * 4;
  float*    deg  = (float*)w;                          w += (size_t)NN * 4;
  float*    dinv = (float*)w;                          w += (size_t)NN * 4;
  float*    sq   = (float*)w;                          w += (size_t)NN * 4;
  float*    pval = (float*)w;                          w += (size_t)NSTRIP * NCH * 16 * KTOP * 4;
  int*      pidx = (int*)w;

  float* hout = (float*)d_out;                         // h occupies first N*C floats

  // GCNConv
  k_conv_x <<<(NN * DD + 255) / 256, 256, 0, stream>>>(x, xbf);
  k_conv_wt<<<(DD * CC + 255) / 256, 256, 0, stream>>>(W, wtbf);
  k_init_deg<<<(NN + 255) / 256, 256, 0, stream>>>(deg);
  k_deg    <<<(EE + 255) / 256, 256, 0, stream>>>(ei, deg);
  k_dinv   <<<(NN + 255) / 256, 256, 0, stream>>>(deg, dinv);
  k_gemm_xw<<<dim3(NSTRIP, CC / 16), 32, 0, stream>>>(xbf, wtbf, h0);
  k_self   <<<(NN * CC + 255) / 256, 256, 0, stream>>>(h0, dinv, b, hout);
  k_edges  <<<20480, 256, 0, stream>>>(ei, h0, dinv, hout);

  // pairwise-distance sampling
  k_sq_bf   <<<NN, 128, 0, stream>>>(hout, hbf, sq);
  k_hht_topk<<<dim3(NSTRIP, NCH), 32, 0, stream>>>(hbf, sq, q, temp, pval, pidx);
  k_merge   <<<(NN + 255) / 256, 256, 0, stream>>>(pval, pidx, (float*)d_out);
}
